// MyGRUCell_37555194036855
// MI455X (gfx1250) — matrix operand order repaired and checked by reading, not compiled, not run on hardware
//
#include <hip/hip_runtime.h>
#include <hip/hip_bf16.h>
#include <math.h>

#define B_DIM   4096
#define H_DIM   1024
#define DX_DIM  512
#define DC_DIM  64
#define KTOT    1600   // DC + DX + H
#define KSPLIT  576    // DC + DX  (h-section starts here)

typedef __attribute__((ext_vector_type(16))) __bf16 v16bf;
typedef __attribute__((ext_vector_type(8)))  __bf16 v8bf;
typedef __attribute__((ext_vector_type(8)))  float  v8f;

union AFrag { v16bf v; v8bf h[2]; };

// ---------------------------------------------------------------------------
// Pack activations [ext | spikes | carry] -> bf16 A[b][k], k-contiguous.
// ---------------------------------------------------------------------------
__global__ void pack_A(const float* __restrict__ carry,
                       const float* __restrict__ spikes,
                       const float* __restrict__ ext,
                       __bf16* __restrict__ A) {
    int k = blockIdx.x * blockDim.x + threadIdx.x;
    int b = blockIdx.y;
    if (k >= KTOT) return;
    float v;
    if (k < DC_DIM)       v = ext[b * DC_DIM + k];
    else if (k < KSPLIT)  v = spikes[b * DX_DIM + (k - DC_DIM)];
    else                  v = carry[b * H_DIM + (k - KSPLIT)];
    A[(size_t)b * KTOT + k] = (__bf16)v;
}

// ---------------------------------------------------------------------------
// Transpose-pack weights -> bf16 Wt[gate][j][k] (K contiguous per column).
// 32x32 LDS tile transpose so both global sides are coalesced.
// ---------------------------------------------------------------------------
__global__ void pack_W(const float* __restrict__ Wirc, const float* __restrict__ Wizc, const float* __restrict__ Winc,
                       const float* __restrict__ Wirx, const float* __restrict__ Wizx, const float* __restrict__ Winx,
                       const float* __restrict__ Whr,  const float* __restrict__ Whz,  const float* __restrict__ Whn,
                       __bf16* __restrict__ Wt) {
    __shared__ __bf16 tile[32][33];
    int k0 = blockIdx.x * 32;
    int j0 = blockIdx.y * 32;
    int g  = blockIdx.z;
    int tx = threadIdx.x & 31;
    int ty = threadIdx.x >> 5;    // 0..7

    const float* src;
    int base;
    if (k0 < DC_DIM)      { src = (g == 0) ? Wirc : (g == 1) ? Wizc : Winc; base = 0; }
    else if (k0 < KSPLIT) { src = (g == 0) ? Wirx : (g == 1) ? Wizx : Winx; base = DC_DIM; }
    else                  { src = (g == 0) ? Whr  : (g == 1) ? Whz  : Whn;  base = KSPLIT; }

    #pragma unroll
    for (int kk = ty; kk < 32; kk += 8)
        tile[kk][tx] = (__bf16)src[(size_t)(k0 - base + kk) * H_DIM + (j0 + tx)];
    __syncthreads();

    __bf16* dst = Wt + (size_t)g * H_DIM * KTOT;
    #pragma unroll
    for (int jj = ty; jj < 32; jj += 8)
        dst[(size_t)(j0 + jj) * KTOT + (k0 + tx)] = tile[tx][jj];
}

// ---------------------------------------------------------------------------
// Fused GRU GEMM + epilogue. Block = 8 waves (wave32), tile 128(M) x 64(N).
// Each wave: 32x32 per gate = 2x2 WMMA tiles, 4 accumulator sets
// (r, z, n_cx for K<576, n_h for K>=576).
// ---------------------------------------------------------------------------
#define K_STEP(KK, ACCN)                                                        \
    {                                                                           \
        AFrag a0, a1;                                                           \
        a0.h[0] = *(const v8bf*)(aRow0 + (KK) + 16 * hi);                       \
        a0.h[1] = *(const v8bf*)(aRow0 + (KK) + 16 * hi + 8);                   \
        a1.h[0] = *(const v8bf*)(aRow1 + (KK) + 16 * hi);                       \
        a1.h[1] = *(const v8bf*)(aRow1 + (KK) + 16 * hi + 8);                   \
        v16bf fR0 = *(const v16bf*)(pR0 + (KK) + 16 * hi);                      \
        v16bf fR1 = *(const v16bf*)(pR1 + (KK) + 16 * hi);                      \
        v16bf fZ0 = *(const v16bf*)(pZ0 + (KK) + 16 * hi);                      \
        v16bf fZ1 = *(const v16bf*)(pZ1 + (KK) + 16 * hi);                      \
        v16bf fN0 = *(const v16bf*)(pN0 + (KK) + 16 * hi);                      \
        v16bf fN1 = *(const v16bf*)(pN1 + (KK) + 16 * hi);                      \
        accR[0][0] = __builtin_amdgcn_wmma_f32_16x16x32_bf16(false, a0.v, false, fR0, (short)0, accR[0][0], false, false); \
        accR[0][1] = __builtin_amdgcn_wmma_f32_16x16x32_bf16(false, a0.v, false, fR1, (short)0, accR[0][1], false, false); \
        accR[1][0] = __builtin_amdgcn_wmma_f32_16x16x32_bf16(false, a1.v, false, fR0, (short)0, accR[1][0], false, false); \
        accR[1][1] = __builtin_amdgcn_wmma_f32_16x16x32_bf16(false, a1.v, false, fR1, (short)0, accR[1][1], false, false); \
        accZ[0][0] = __builtin_amdgcn_wmma_f32_16x16x32_bf16(false, a0.v, false, fZ0, (short)0, accZ[0][0], false, false); \
        accZ[0][1] = __builtin_amdgcn_wmma_f32_16x16x32_bf16(false, a0.v, false, fZ1, (short)0, accZ[0][1], false, false); \
        accZ[1][0] = __builtin_amdgcn_wmma_f32_16x16x32_bf16(false, a1.v, false, fZ0, (short)0, accZ[1][0], false, false); \
        accZ[1][1] = __builtin_amdgcn_wmma_f32_16x16x32_bf16(false, a1.v, false, fZ1, (short)0, accZ[1][1], false, false); \
        ACCN[0][0] = __builtin_amdgcn_wmma_f32_16x16x32_bf16(false, a0.v, false, fN0, (short)0, ACCN[0][0], false, false); \
        ACCN[0][1] = __builtin_amdgcn_wmma_f32_16x16x32_bf16(false, a0.v, false, fN1, (short)0, ACCN[0][1], false, false); \
        ACCN[1][0] = __builtin_amdgcn_wmma_f32_16x16x32_bf16(false, a1.v, false, fN0, (short)0, ACCN[1][0], false, false); \
        ACCN[1][1] = __builtin_amdgcn_wmma_f32_16x16x32_bf16(false, a1.v, false, fN1, (short)0, ACCN[1][1], false, false); \
    }

__global__ __launch_bounds__(256)
void gru_gemm(const __bf16* __restrict__ A,
              const __bf16* __restrict__ Wt,
              const float* __restrict__ carry,
              const float* __restrict__ br,
              const float* __restrict__ bz,
              const float* __restrict__ bn,
              float* __restrict__ out) {
    const int lane = threadIdx.x & 31;
    const int wave = threadIdx.x >> 5;
    const int wm   = wave & 3;                       // M quadrant within block
    const int wn   = wave >> 2;                      // N half within block
    const int rowBase = blockIdx.x * 128 + wm * 32;
    const int colBase = blockIdx.y * 64  + wn * 32;
    const int hi = lane >> 4;
    const int lm = lane & 15;

    v8f accR[2][2]   = {};
    v8f accZ[2][2]   = {};
    v8f accNcx[2][2] = {};
    v8f accNh[2][2]  = {};

    // A fragment row pointers (lane L -> M = L%16; 16 contiguous K per half, k = 16*(L/16) + s)
    const __bf16* aRow0 = A + (size_t)(rowBase + lm) * KTOT;
    const __bf16* aRow1 = A + (size_t)(rowBase + 16 + lm) * KTOT;
    // B fragment column pointers (lane L -> N = L%16; 16 contiguous K per half)
    const __bf16* pR0 = Wt + ((size_t)0 * H_DIM + colBase +      lm) * KTOT;
    const __bf16* pR1 = Wt + ((size_t)0 * H_DIM + colBase + 16 + lm) * KTOT;
    const __bf16* pZ0 = Wt + ((size_t)1 * H_DIM + colBase +      lm) * KTOT;
    const __bf16* pZ1 = Wt + ((size_t)1 * H_DIM + colBase + 16 + lm) * KTOT;
    const __bf16* pN0 = Wt + ((size_t)2 * H_DIM + colBase +      lm) * KTOT;
    const __bf16* pN1 = Wt + ((size_t)2 * H_DIM + colBase + 16 + lm) * KTOT;

    // Phase 1: K in [0, 576)  -> n-gate goes to accNcx (ext+spikes part)
    for (int kk = 0; kk < KSPLIT; kk += 32) K_STEP(kk, accNcx);
    // Phase 2: K in [576,1600) -> n-gate goes to accNh (h part; gets r-scaled)
    for (int kk = KSPLIT; kk < KTOT; kk += 32) K_STEP(kk, accNh);

    // Fused GRU epilogue. C/D layout: element r of v8f -> M = r + 8*hi, N = lm.
    #pragma unroll
    for (int mi = 0; mi < 2; ++mi) {
        #pragma unroll
        for (int ni = 0; ni < 2; ++ni) {
            const int col = colBase + ni * 16 + lm;
            const float brv = br[col];
            const float bzv = bz[col];
            const float bnv = bn[col];
            #pragma unroll
            for (int r = 0; r < 8; ++r) {
                const int row = rowBase + mi * 16 + r + 8 * hi;
                const float rv = accR[mi][ni][r] + brv;
                const float zv = accZ[mi][ni][r] + bzv;
                const float nh = accNh[mi][ni][r] + bnv;      // h@Whn + bn
                const float rs = 1.0f / (1.0f + __expf(-rv));
                const float zs = 1.0f / (1.0f + __expf(-zv));
                const float nt = tanhf(accNcx[mi][ni][r] + rs * nh);
                const float h0 = carry[(size_t)row * H_DIM + col];
                const float hn = (1.0f - zs) * nt + zs * h0;
                out[(size_t)row * H_DIM + col] = hn;
                out[(size_t)B_DIM * H_DIM + (size_t)row * H_DIM + col] = hn;
            }
        }
    }
}

extern "C" void kernel_launch(void* const* d_in, const int* in_sizes, int n_in,
                              void* d_out, int out_size, void* d_ws, size_t ws_size,
                              hipStream_t stream) {
    const float* carry  = (const float*)d_in[0];
    const float* spikes = (const float*)d_in[1];
    const float* ext    = (const float*)d_in[2];
    const float* Wirc   = (const float*)d_in[3];
    const float* Wirx   = (const float*)d_in[4];
    const float* Wizc   = (const float*)d_in[5];
    const float* Wizx   = (const float*)d_in[6];
    const float* Winc   = (const float*)d_in[7];
    const float* Winx   = (const float*)d_in[8];
    const float* Whr    = (const float*)d_in[9];
    const float* br     = (const float*)d_in[10];
    const float* Whz    = (const float*)d_in[11];
    const float* bz     = (const float*)d_in[12];
    const float* Whn    = (const float*)d_in[13];
    const float* bn     = (const float*)d_in[14];

    __bf16* Abf = (__bf16*)d_ws;                            // 4096*1600 bf16 = 13.1 MB
    __bf16* Wtb = Abf + (size_t)B_DIM * KTOT;               // 3*1024*1600 bf16 = 9.8 MB

    pack_A<<<dim3((KTOT + 255) / 256, B_DIM), 256, 0, stream>>>(carry, spikes, ext, Abf);
    pack_W<<<dim3(KTOT / 32, H_DIM / 32, 3), 256, 0, stream>>>(
        Wirc, Wizc, Winc, Wirx, Wizx, Winx, Whr, Whz, Whn, Wtb);
    gru_gemm<<<dim3(B_DIM / 128, H_DIM / 64), 256, 0, stream>>>(
        Abf, Wtb, carry, br, bz, bn, (float*)d_out);
}
